// Encoder_30545807409172
// MI455X (gfx1250) — matrix operand order repaired and checked by reading, not compiled, not run on hardware
//
#include <hip/hip_runtime.h>
#include <hip/hip_bf16.h>

// ---------------------------------------------------------------------------
// GAT encoder for MI455X (gfx1250, wave32).
//   - attention GEMM [E,192]x[192,64]  -> v_wmma_f32_16x16x32_bf16
//   - node transform [N,64]x[64,64]    -> v_wmma_f32_16x16x4_f32 (full fp32)
//   - softmax / scatter passes vectorized x4, segment-max via ordered-uint
// ---------------------------------------------------------------------------

#define NN   50000
#define EE   800000
#define HH   64
#define KK   192          // 2H + HE
#define NEG_SLOPE 0.2f
#define BN_EPS    1e-5f

typedef __attribute__((ext_vector_type(16))) __bf16 v16bf;
typedef __attribute__((ext_vector_type(8)))  float  v8f;
typedef __attribute__((ext_vector_type(2)))  float  v2f;

// float -> bf16 round-to-nearest-even (bit-level)
__device__ __forceinline__ unsigned short f2bf(float f) {
    unsigned u = __float_as_uint(f);
    u += 0x7fffu + ((u >> 16) & 1u);
    return (unsigned short)(u >> 16);
}

// order-preserving float <-> uint encoding for atomicMax segment-max
__device__ __forceinline__ unsigned fenc(float f) {
    unsigned u = __float_as_uint(f);
    return (u & 0x80000000u) ? ~u : (u | 0x80000000u);
}
__device__ __forceinline__ float fdec(unsigned u) {
    unsigned b = (u & 0x80000000u) ? (u ^ 0x80000000u) : ~u;
    return __uint_as_float(b);
}
#define ENC_NEG_INF 0x007FFFFFu   // fenc(-inf)

// ---------------------------------------------------------------------------
// K1: h_pre = x @ fc_node_w + b ; accumulate per-channel sum / sumsq
// ---------------------------------------------------------------------------
__global__ void __launch_bounds__(256)
node_linear_stats(const float* __restrict__ x, const float* __restrict__ w,
                  const float* __restrict__ b, float* __restrict__ hpre,
                  float* __restrict__ gsum, float* __restrict__ gsq, int n) {
    const int c = threadIdx.x & 63;
    const int g = threadIdx.x >> 6;              // 4 rows per block
    const float w0 = w[c], w1 = w[64 + c], w2 = w[128 + c], bc = b[c];
    float ls = 0.f, lq = 0.f;
    for (int row = blockIdx.x * 4 + g; row < n; row += gridDim.x * 4) {
        const float* xr = x + row * 3;
        float v = xr[0] * w0 + xr[1] * w1 + xr[2] * w2 + bc;
        hpre[row * 64 + c] = v;
        ls += v; lq += v * v;
    }
    __shared__ float ssum[4][64], ssq[4][64];
    ssum[g][c] = ls; ssq[g][c] = lq;
    __syncthreads();
    if (g == 0) {
        float S = ssum[0][c] + ssum[1][c] + ssum[2][c] + ssum[3][c];
        float Q = ssq[0][c] + ssq[1][c] + ssq[2][c] + ssq[3][c];
        atomicAdd(&gsum[c], S);
        atomicAdd(&gsq[c], Q);
    }
}

// K2: per-channel BN scale/shift (1 block, 64 threads)
__global__ void bn_finalize(const float* __restrict__ gsum, const float* __restrict__ gsq,
                            const float* __restrict__ gamma, const float* __restrict__ beta,
                            float* __restrict__ scale, float* __restrict__ shift, int n) {
    int c = threadIdx.x;
    float m = gsum[c] / (float)n;
    float v = gsq[c] / (float)n - m * m;
    float s = gamma[c] * rsqrtf(v + BN_EPS);
    scale[c] = s;
    shift[c] = beta[c] - m * s;
}

// K3: h = h_pre * scale + shift   (x4 vectorized; total = N*64/4 threads)
__global__ void __launch_bounds__(256)
bn_apply4(const float* __restrict__ hpre, const float* __restrict__ scale,
          const float* __restrict__ shift, float* __restrict__ h, int total4) {
    int i = blockIdx.x * 256 + threadIdx.x;
    if (i >= total4) return;
    int c4 = (i & 15) << 2;
    float4 hp = *(const float4*)(hpre + i * 4);
    float4 sc = *(const float4*)(scale + c4);
    float4 sh = *(const float4*)(shift + c4);
    float4 o;
    o.x = hp.x * sc.x + sh.x;  o.y = hp.y * sc.y + sh.y;
    o.z = hp.z * sc.z + sh.z;  o.w = hp.w * sc.w + sh.w;
    *(float4*)(h + i * 4) = o;
}

// K4: e_bf = bf16(edge_attr @ fc_edge_w + b)  (x4 vectorized)
__global__ void __launch_bounds__(256)
edge_linear4(const float* __restrict__ ea, const float* __restrict__ w,
             const float* __restrict__ b, unsigned short* __restrict__ ebf, int nE) {
    __shared__ float sw[256], sb[64];
    if (threadIdx.x < 64) sb[threadIdx.x] = b[threadIdx.x];
    sw[threadIdx.x] = w[threadIdx.x];
    __syncthreads();
    int i = blockIdx.x * 256 + threadIdx.x;
    if (i >= nE * 16) return;
    int e = i >> 4, c4 = (i & 15) << 2;
    float4 r = *(const float4*)(ea + e * 4);
    ushort4 o;
    {
        float v0 = r.x * sw[c4+0] + r.y * sw[64+c4+0] + r.z * sw[128+c4+0] + r.w * sw[192+c4+0] + sb[c4+0];
        float v1 = r.x * sw[c4+1] + r.y * sw[64+c4+1] + r.z * sw[128+c4+1] + r.w * sw[192+c4+1] + sb[c4+1];
        float v2 = r.x * sw[c4+2] + r.y * sw[64+c4+2] + r.z * sw[128+c4+2] + r.w * sw[192+c4+2] + sb[c4+2];
        float v3 = r.x * sw[c4+3] + r.y * sw[64+c4+3] + r.z * sw[128+c4+3] + r.w * sw[192+c4+3] + sb[c4+3];
        o.x = f2bf(v0); o.y = f2bf(v1); o.z = f2bf(v2); o.w = f2bf(v3);
    }
    *(ushort4*)(ebf + e * 64 + c4) = o;
}

// K5: pack conv_attn_w into per-lane-contiguous bf16 WMMA B-fragments.
// Layout: [layer][ks(6)][nt(4)][lane(32)][j(16)]
//   lane<16 : K = 32*ks + j      , N = 16*nt + lane
//   lane>=16: K = 32*ks + 16 + j , N = 16*nt + (lane-16)
__global__ void __launch_bounds__(256)
pack_attn_w(const float* __restrict__ w, unsigned short* __restrict__ wp, int total) {
    int idx = blockIdx.x * 256 + threadIdx.x;
    if (idx >= total) return;
    int j    = idx & 15;
    int lane = (idx >> 4) & 31;
    int nt   = (idx >> 9) & 3;
    int ks   = (idx >> 11) % 6;
    int l    = idx / 12288;
    int k = 32 * ks + ((lane >= 16) ? 16 : 0) + j;
    int n = 16 * nt + (lane & 15);
    wp[idx] = f2bf(w[(l * KK + k) * 64 + n]);
}

// ---------------------------------------------------------------------------
// K6: hl = h @ conv_fc_w[l] + b  on the fp32 matrix pipe.
// One wave = 16 rows x all 64 cols; K=64 -> 16 x v_wmma_f32_16x16x4_f32 per
// N-tile. Weights repacked in LDS as per-lane v2f fragments.
//   A 16x4 f32: lanes 0-15 M=lane, K={0,1}; lanes 16-31 K={2,3}
//   B  4x16 f32: lanes 0-15 N=lane, K={0,1}; lanes 16-31 K={2,3}
// ---------------------------------------------------------------------------
#if __has_builtin(__builtin_amdgcn_wmma_f32_16x16x4_f32)
#define HAVE_WMMA_F32 1
__global__ void __launch_bounds__(256)
hl_wmma(const float* __restrict__ h, const float* __restrict__ w,
        const float* __restrict__ b, float* __restrict__ hl,
        unsigned short* __restrict__ hlbf, int n) {
    __shared__ float swp[4096];   // [ks(16)][nt(4)][lane(32)][j(2)]
    for (int p = threadIdx.x; p < 4096; p += 256) {
        int j    = p & 1;
        int lane = (p >> 1) & 31;
        int nt   = (p >> 6) & 3;
        int ks   = p >> 8;
        int k = 4 * ks + ((lane >= 16) ? 2 : 0) + j;
        int nn = 16 * nt + (lane & 15);
        swp[p] = w[k * 64 + nn];
    }
    __syncthreads();

    const int lane = threadIdx.x & 31;
    const int half = lane >> 4;
    const int l15  = lane & 15;
    const int tile = blockIdx.x * 8 + (threadIdx.x >> 5);
    if (tile * 16 >= n) return;

    const float* hrow = h + (tile * 16 + l15) * 64;

    v8f acc[4];
    #pragma unroll
    for (int nt = 0; nt < 4; ++nt) acc[nt] = (v8f){0.f,0.f,0.f,0.f,0.f,0.f,0.f,0.f};

    #pragma unroll
    for (int ks = 0; ks < 16; ++ks) {
        v2f av = *(const v2f*)(hrow + 4 * ks + 2 * half);
        #pragma unroll
        for (int nt = 0; nt < 4; ++nt) {
            v2f bv = *(const v2f*)(swp + ks * 256 + nt * 64 + lane * 2);
            acc[nt] = __builtin_amdgcn_wmma_f32_16x16x4_f32(
                          false, av, false, bv, (short)0, acc[nt], false, false);
        }
    }

    #pragma unroll
    for (int nt = 0; nt < 4; ++nt) {
        const int nn = nt * 16 + l15;
        const float bias = b[nn];
        #pragma unroll
        for (int r = 0; r < 8; ++r) {
            const int row = tile * 16 + r + 8 * half;
            float v = acc[nt][r] + bias;
            hl[row * 64 + nn] = v;
            hlbf[row * 64 + nn] = f2bf(v);
        }
    }
}
#else
#define HAVE_WMMA_F32 0
__global__ void __launch_bounds__(256)
hl_valu(const float* __restrict__ h, const float* __restrict__ w,
        const float* __restrict__ b, float* __restrict__ hl,
        unsigned short* __restrict__ hlbf, int n) {
    __shared__ float sw[4096];
    __shared__ float sh[4][64];
    for (int i = threadIdx.x; i < 4096; i += 256) sw[i] = w[i];
    const int c = threadIdx.x & 63;
    const int g = threadIdx.x >> 6;
    const int row = blockIdx.x * 4 + g;
    if (row < n) sh[g][c] = h[row * 64 + c];
    __syncthreads();
    if (row >= n) return;
    float acc = b[c];
    #pragma unroll
    for (int k = 0; k < 64; ++k) acc += sh[g][k] * sw[k * 64 + c];
    hl[row * 64 + c] = acc;
    hlbf[row * 64 + c] = f2bf(acc);
}
#endif

// K7: per-layer init of amax/denom/h_next (x4 vectorized)
__global__ void __launch_bounds__(256)
init_layer4(unsigned* __restrict__ amax, float* __restrict__ denom,
            float* __restrict__ hnext, int total4) {
    int i = blockIdx.x * 256 + threadIdx.x;
    if (i >= total4) return;
    uint4 m = make_uint4(ENC_NEG_INF, ENC_NEG_INF, ENC_NEG_INF, ENC_NEG_INF);
    float4 z = make_float4(0.f, 0.f, 0.f, 0.f);
    *(uint4*)(amax + i * 4)   = m;
    *(float4*)(denom + i * 4) = z;
    *(float4*)(hnext + i * 4) = z;
}

// ---------------------------------------------------------------------------
// K8: fused gather + WMMA attention GEMM + bias + leaky-relu + segment-max.
// One wave = 16 edges x 16 channels; K=192 -> 6 x v_wmma_f32_16x16x32_bf16.
// ---------------------------------------------------------------------------
__global__ void __launch_bounds__(256)
attn_wmma(const unsigned short* __restrict__ hlbf,   // [N,64] bf16
          const unsigned short* __restrict__ ebf,    // [E,64] bf16
          const unsigned short* __restrict__ wpack,  // layer slice [6][4][32][16]
          const float* __restrict__ attn_b,          // [64] layer slice
          const long long* __restrict__ ei,          // [2,E]
          float* __restrict__ alpha,                 // [E,64]
          unsigned* __restrict__ amax,               // [N,64] encoded
          int nE) {
    const int lane = threadIdx.x & 31;
    const int wid  = blockIdx.x * (blockDim.x >> 5) + (threadIdx.x >> 5);
    const int nWaves = gridDim.x * (blockDim.x >> 5);
    const int nt   = wid & 3;
    const int half = lane >> 4;
    const int l15  = lane & 15;

    // Resident B fragments for this 16-channel tile
    v16bf Breg[6];
    #pragma unroll
    for (int ks = 0; ks < 6; ++ks) {
        const uint4* bp = reinterpret_cast<const uint4*>(wpack + ((ks * 4 + nt) * 32 + lane) * 16);
        union { v16bf v; uint4 q[2]; } u;
        u.q[0] = bp[0]; u.q[1] = bp[1];
        Breg[ks] = u.v;
    }

    const int   nGlobal = nt * 16 + l15;
    const float bias = attn_b[nGlobal];
    const int   nTiles = nE >> 4;

    for (int et = wid >> 2; et < nTiles; et += (nWaves >> 2)) {
        const int e0 = et * 16;
        const int eidA = e0 + l15;                 // A-row this lane feeds
        const int srcN = (int)ei[eidA];
        const int dstN = (int)ei[nE + eidA];
        const unsigned short* xi = hlbf + dstN * 64;   // x_i = hl[dst]
        const unsigned short* xj = hlbf + srcN * 64;   // x_j = hl[src]
        const unsigned short* ee = ebf + eidA * 64;

        v8f acc = {0.f, 0.f, 0.f, 0.f, 0.f, 0.f, 0.f, 0.f};
        #pragma unroll
        for (int ks = 0; ks < 6; ++ks) {
            // A layout: lanes 0-15 hold K = 32ks+[0..15];
            //           lanes 16-31 hold K = 32ks+[16..31]
            const int k0 = 32 * ks + 16 * half;
            const int k1 = k0 + 8;
            const unsigned short* s0 = (k0 < 64) ? (xi + k0) : (k0 < 128) ? (xj + k0 - 64) : (ee + k0 - 128);
            const unsigned short* s1 = (k1 < 64) ? (xi + k1) : (k1 < 128) ? (xj + k1 - 64) : (ee + k1 - 128);
            union { v16bf v; uint4 q[2]; } a;
            a.q[0] = *reinterpret_cast<const uint4*>(s0);
            a.q[1] = *reinterpret_cast<const uint4*>(s1);
            acc = __builtin_amdgcn_wmma_f32_16x16x32_bf16(
                      false, a.v, false, Breg[ks], (short)0, acc, false, false);
        }

        // D layout: VGPR r, lanes 0-15 -> M=r, lanes 16-31 -> M=r+8, N=lane&15
        // dst indices for all 16 rows already live in the wave: lane m holds
        // dst of edge e0+m -> broadcast with wave32 shuffle instead of reloads.
        #pragma unroll
        for (int r = 0; r < 8; ++r) {
            const int m   = r + 8 * half;
            const int eid = e0 + m;
            float v = acc[r] + bias;
            v = (v > 0.f) ? v : NEG_SLOPE * v;
            alpha[eid * 64 + nGlobal] = v;
            const int d = __shfl(dstN, m);
            atomicMax(amax + d * 64 + nGlobal, fenc(v));
        }
    }
}

// K9: alpha = exp(alpha - amax[dst]); denom[dst] += alpha   (x4 vectorized)
__global__ void __launch_bounds__(256)
softmax_exp4(float* __restrict__ alpha, const unsigned* __restrict__ amax,
             float* __restrict__ denom, const long long* __restrict__ ei, int nE) {
    int i = blockIdx.x * 256 + threadIdx.x;
    if (i >= nE * 16) return;
    int e = i >> 4, c4 = (i & 15) << 2;
    int d = (int)ei[nE + e];
    float4 a = *(const float4*)(alpha + e * 64 + c4);
    uint4  m = *(const uint4*)(amax + d * 64 + c4);
    float4 p;
    p.x = __expf(a.x - fdec(m.x));
    p.y = __expf(a.y - fdec(m.y));
    p.z = __expf(a.z - fdec(m.z));
    p.w = __expf(a.w - fdec(m.w));
    *(float4*)(alpha + e * 64 + c4) = p;
    float* dn = denom + d * 64 + c4;
    atomicAdd(dn + 0, p.x);
    atomicAdd(dn + 1, p.y);
    atomicAdd(dn + 2, p.z);
    atomicAdd(dn + 3, p.w);
}

// K10: h_next[dst] += hl[src] * alpha / (denom[dst] + eps)   (x4 vectorized)
__global__ void __launch_bounds__(256)
aggregate4(const float* __restrict__ alpha, const float* __restrict__ denom,
           const float* __restrict__ hl, const long long* __restrict__ ei,
           float* __restrict__ hnext, int nE) {
    int i = blockIdx.x * 256 + threadIdx.x;
    if (i >= nE * 16) return;
    int e = i >> 4, c4 = (i & 15) << 2;
    int d = (int)ei[nE + e];
    int s = (int)ei[e];
    float4 a  = *(const float4*)(alpha + e * 64 + c4);
    float4 dn = *(const float4*)(denom + d * 64 + c4);
    float4 hv = *(const float4*)(hl + s * 64 + c4);
    float* out = hnext + d * 64 + c4;
    atomicAdd(out + 0, hv.x * (a.x / (dn.x + 1e-16f)));
    atomicAdd(out + 1, hv.y * (a.y / (dn.y + 1e-16f)));
    atomicAdd(out + 2, hv.z * (a.z / (dn.z + 1e-16f)));
    atomicAdd(out + 3, hv.w * (a.w / (dn.w + 1e-16f)));
}

__global__ void fill_f32(float* p, int n, float v) {
    int i = blockIdx.x * 256 + threadIdx.x;
    if (i < n) p[i] = v;
}

// ---------------------------------------------------------------------------
extern "C" void kernel_launch(void* const* d_in, const int* in_sizes, int n_in,
                              void* d_out, int out_size, void* d_ws, size_t ws_size,
                              hipStream_t stream) {
    const float*     x     = (const float*)d_in[0];
    const long long* ei    = (const long long*)d_in[1];
    const float*     ea    = (const float*)d_in[2];
    const float*     fnw   = (const float*)d_in[3];
    const float*     fnb   = (const float*)d_in[4];
    const float*     few   = (const float*)d_in[5];
    const float*     feb   = (const float*)d_in[6];
    const float*     gamma = (const float*)d_in[7];
    const float*     beta  = (const float*)d_in[8];
    const float*     cfw   = (const float*)d_in[9];
    const float*     cfb   = (const float*)d_in[10];
    const float*     caw   = (const float*)d_in[11];
    const float*     cab   = (const float*)d_in[12];

    // workspace carve (256 B aligned)
    char* base = (char*)d_ws;
    size_t off = 0;
    auto carve = [&](size_t bytes) -> void* {
        void* p = base + off;
        off += (bytes + 255) & ~(size_t)255;
        return p;
    };
    float*          hlF   = (float*)carve((size_t)NN * 64 * 4);   // hl fp32 (also h_pre temp)
    float*          hA    = (float*)carve((size_t)NN * 64 * 4);   // h ping
    float*          hB    = (float*)carve((size_t)NN * 64 * 4);   // h pong
    unsigned short* hlBF  = (unsigned short*)carve((size_t)NN * 64 * 2);
    unsigned short* eBF   = (unsigned short*)carve((size_t)EE * 64 * 2);
    float*          alpha = (float*)carve((size_t)EE * 64 * 4);
    unsigned*       amax  = (unsigned*)carve((size_t)NN * 64 * 4);
    float*          denom = (float*)carve((size_t)NN * 64 * 4);
    unsigned short* wpack = (unsigned short*)carve((size_t)3 * 12288 * 2);
    float*          stats = (float*)carve(4 * 64 * 4);  // sum | sumsq | scale | shift

    const int nodeTot4 = NN * 16;      //   800,000 (x4-vector work items)
    const int edgeTot4 = EE * 16;      // 12,800,000
    const int nodeBlk4 = (nodeTot4 + 255) / 256;   // 3125
    const int edgeBlk4 = (edgeTot4 + 255) / 256;   // 50000

    // ---- input embedding + BatchNorm -------------------------------------
    fill_f32<<<1, 256, 0, stream>>>(stats, 128, 0.f);
    node_linear_stats<<<512, 256, 0, stream>>>(x, fnw, fnb, hlF, stats, stats + 64, NN);
    bn_finalize<<<1, 64, 0, stream>>>(stats, stats + 64, gamma, beta, stats + 128, stats + 192, NN);
    bn_apply4<<<nodeBlk4, 256, 0, stream>>>(hlF, stats + 128, stats + 192, hA, nodeTot4);

    // ---- edge embedding (bf16) + packed WMMA weights ---------------------
    edge_linear4<<<edgeBlk4, 256, 0, stream>>>(ea, few, feb, eBF, EE);
    pack_attn_w<<<(3 * 12288 + 255) / 256, 256, 0, stream>>>(caw, wpack, 3 * 12288);

    // ---- 3 GAT layers ----------------------------------------------------
    float* hcur = hA;
    for (int l = 0; l < 3; ++l) {
        float* hnext = (l == 2) ? (float*)d_out : ((l == 0) ? hB : hA);

#if HAVE_WMMA_F32
        hl_wmma<<<(NN / 16 + 7) / 8, 256, 0, stream>>>(hcur, cfw + l * 64 * 64, cfb + l * 64,
                                                       hlF, hlBF, NN);
#else
        hl_valu<<<(NN + 3) / 4, 256, 0, stream>>>(hcur, cfw + l * 64 * 64, cfb + l * 64,
                                                  hlF, hlBF, NN);
#endif
        init_layer4<<<nodeBlk4, 256, 0, stream>>>(amax, denom, hnext, nodeTot4);
        attn_wmma<<<2500, 256, 0, stream>>>(hlBF, eBF, wpack + l * 12288, cab + l * 64,
                                            ei, alpha, amax, EE);
        softmax_exp4<<<edgeBlk4, 256, 0, stream>>>(alpha, amax, denom, ei, EE);
        aggregate4<<<edgeBlk4, 256, 0, stream>>>(alpha, denom, hlF, ei, hnext, EE);

        hcur = hnext;
    }
    (void)in_sizes; (void)n_in; (void)out_size; (void)ws_size;
}
